// HashEncoding_SDF_111669150287
// MI455X (gfx1250) — matrix operand order repaired and checked by reading, not compiled, not run on hardware
//
#include <hip/hip_runtime.h>

typedef __attribute__((ext_vector_type(16))) _Float16 v16h;
typedef __attribute__((ext_vector_type(8)))  _Float16 v8h;
typedef __attribute__((ext_vector_type(2)))  _Float16 h2;
typedef __attribute__((ext_vector_type(8)))  float    v8f;
typedef __attribute__((ext_vector_type(4)))  float    v4f;
typedef __attribute__((ext_vector_type(8)))  unsigned v8u;
typedef __attribute__((ext_vector_type(4)))  unsigned v4u;

#define LVLS      16
#define THASH     (1u << 19)
#define TILE_PTS  256               // 8 waves x 32 points (2 column-tiles per wave)
#define NWAVES    8

// ---- LDS layout ----
// Weights pre-swizzled as WMMA A-fragments of W^T (rows = out feats, cols = in feats).
// Fragment = 32 lanes x 16 halves = 512 halves, stored [nt][kt] per layer.
// Layer 0 bakes swap8 (bit3<->bit4) into k; layers 1..4 keep table order so D->B needs no shuffles.
#define W0_OFF 0                      // 8 nt * 1 kt
#define W1_OFF 4096                   // 8 nt * 4 kt
#define W2_OFF 20480
#define W3_OFF 36864
#define W4_OFF 53248                  // 1 nt * 4 kt (out 4 -> padded 16 rows)
#define WTOT_HALVES 55296
#define BIASH_OFF_BYTES (WTOT_HALVES * 2)             // 110592: 512 f16 (b0..b3)
#define BIAS4_OFF_BYTES (BIASH_OFF_BYTES + 512 * 2)   // 111616: 16 f32 (b4 padded)
#define ENC_OFF_BYTES   (BIAS4_OFF_BYTES + 16 * 4)    // 111680; enc: 8 waves * 32 pts * 32 halves
#define LDS_BYTES       (ENC_OFF_BYTES + NWAVES * 32 * 32 * 2)  // 128064 B -> 2 blocks/WGP

__device__ __forceinline__ v16h load_w_frag(const _Float16* frag, int lane) {
  const _Float16* p = frag + lane * 16;
  v8h lo = *reinterpret_cast<const v8h*>(p);
  v8h hi = *reinterpret_cast<const v8h*>(p + 8);
  v16h a;
#pragma unroll
  for (int i = 0; i < 8; ++i) { a[i] = lo[i]; a[i + 8] = hi[i]; }
  return a;
}

// f16 C/D (transposed MLP): col = point = lane%16, row/feature = 8*(lane/16) + 2r + s.
// Bias pairs are 8 consecutive halves per lane: one ds_load_b128 IS the C fragment.
__device__ __forceinline__ v8h bias_frag_h(const _Float16* blh, int nt, int hh) {
  return *reinterpret_cast<const v8h*>(blh + nt * 16 + hh * 8);
}

// f32 C/D for the final layer: 8 consecutive floats per lane.
__device__ __forceinline__ v8f bias_frag8(const float* bl, int nt, int hh) {
  const float* bp = bl + nt * 16 + hh * 8;
  v4f x0 = *reinterpret_cast<const v4f*>(bp);
  v4f x1 = *reinterpret_cast<const v4f*>(bp + 4);
  v8f c;
#pragma unroll
  for (int i = 0; i < 4; ++i) { c[i] = x0[i]; c[i + 4] = x1[i]; }
  return c;
}

// Packed ReLU in one op per f16x2 (inline asm avoids LLVM canonicalize-max).
__device__ __forceinline__ unsigned pk_relu_u(unsigned u) {
  unsigned r;
  asm("v_pk_max_num_f16 %0, %1, 0" : "=v"(r) : "v"(u));
  return r;
}

// f16 D tiles are already packed in ascending-feature pairs (M = 8h+2r+s):
// D(2kt) -> elements 0..7, D(2kt+1) -> elements 8..15. Pure register concat.
__device__ __forceinline__ v16h pack2_relu_h(v8h d0, v8h d1) {
  v4u u0 = __builtin_bit_cast(v4u, d0);
  v4u u1 = __builtin_bit_cast(v4u, d1);
  v8u u;
#pragma unroll
  for (int i = 0; i < 4; ++i) {
    u[i]     = pk_relu_u(u0[i]);
    u[i + 4] = pk_relu_u(u1[i]);
  }
  return __builtin_bit_cast(v16h, u);
}

__device__ __forceinline__ v8h wmma_acc_h(v16h a, v16h b, v8h c) {
  return __builtin_amdgcn_wmma_f16_16x16x32_f16(false, a, false, b, (short)0, c, false, false);
}
__device__ __forceinline__ v8f wmma_acc_f(v16h a, v16h b, v8f c) {
  return __builtin_amdgcn_wmma_f32_16x16x32_f16(false, a, false, b, (short)0, c, false, false);
}

__global__ void __launch_bounds__(256)
ngp_fused_kernel(const float* __restrict__ x, const float* __restrict__ table,
                 const float* __restrict__ W0, const float* __restrict__ b0,
                 const float* __restrict__ W1, const float* __restrict__ b1,
                 const float* __restrict__ W2, const float* __restrict__ b2,
                 const float* __restrict__ W3, const float* __restrict__ b3,
                 const float* __restrict__ W4, const float* __restrict__ b4,
                 float* __restrict__ out, int N, int nTiles) {
  extern __shared__ char smem[];
  _Float16* wLds   = reinterpret_cast<_Float16*>(smem);
  _Float16* biasH  = reinterpret_cast<_Float16*>(smem + BIASH_OFF_BYTES);
  float*    bias4  = reinterpret_cast<float*>(smem + BIAS4_OFF_BYTES);
  _Float16* encLds = reinterpret_cast<_Float16*>(smem + ENC_OFF_BYTES);

  const int tid  = threadIdx.x;
  const int lane = tid & 31;
  const int wave = tid >> 5;
  const int col  = lane & 15;
  const int hh   = lane >> 4;

  // ---- one-time: swizzle fp32 weights -> f16 A-fragments in LDS ----
  {
    const float* Ws[5] = {W0, W1, W2, W3, W4};
    const int kin[5]   = {32, 128, 128, 128, 128};
    const int nout[5]  = {128, 128, 128, 128, 4};
    const int offs[5]  = {W0_OFF, W1_OFF, W2_OFF, W3_OFF, W4_OFF};
#pragma unroll
    for (int l = 0; l < 5; ++l) {
      const int ktn = kin[l] >> 5;
      const int ntn = (nout[l] + 15) >> 4;
      const int tot = ktn * ntn * 512;
      const float* W = Ws[l];
      for (int i = tid; i < tot; i += 256) {
        int frag = i >> 9, rem = i & 511;
        int ln = rem >> 4, e = rem & 15;
        int nt = frag / ktn, kt = frag - nt * ktn;
        int n = nt * 16 + (ln & 15);
        int kpos = (e >> 3) * 16 + (ln >> 4) * 8 + (e & 7);   // A layout per 7.12.2
        int kk = (l != 0) ? kpos
                          : ((kpos & 7) | ((kpos & 8) << 1) | ((kpos & 16) >> 1));
        int k = kt * 32 + kk;
        float v = (n < nout[l]) ? W[k * nout[l] + n] : 0.0f;
        wLds[offs[l] + i] = (_Float16)v;
      }
    }
    const float* bs[4] = {b0, b1, b2, b3};
    for (int i = tid; i < 512; i += 256) biasH[i] = (_Float16)bs[i >> 7][i & 127];
    if (tid < 16) bias4[tid] = (tid < 4) ? b4[tid] : 0.0f;
  }

  _Float16* encW = encLds + wave * (32 * 32);

  for (int tile = blockIdx.x; tile < nTiles; tile += gridDim.x) {
    const int base = tile * TILE_PTS;
    __syncthreads();

    // ---- hash-grid encoding: one point per thread, 16 levels (table L2-resident) ----
    {
      const int p = tid;
      const int gp = base + p;
      if (gp < N) {
        float xx = x[gp * 3 + 0], yy = x[gp * 3 + 1], zz = x[gp * 3 + 2];
        _Float16* encRow = encLds + (p >> 5) * 1024 + (p & 31) * 32;
#pragma unroll 4
        for (int lvl = 0; lvl < LVLS; ++lvl) {
          float r = exp2f(4.0f + (7.0f / 15.0f) * (float)lvl);   // 16 * 128^(lvl/15)
          float px = xx * r, py = yy * r, pz = zz * r;
          float fx = floorf(px), fy = floorf(py), fz = floorf(pz);
          float wx = px - fx, wy = py - fy, wz = pz - fz;
          unsigned ix = (unsigned)(int)fx, iy = (unsigned)(int)fy, iz = (unsigned)(int)fz;
          unsigned hx[2] = { ix, ix + 1u };
          unsigned hy[2] = { iy * 2654435761u, (iy + 1u) * 2654435761u };
          unsigned hz[2] = { iz * 805459861u, (iz + 1u) * 805459861u };
          float wxa[2] = {1.f - wx, wx}, wya[2] = {1.f - wy, wy}, wza[2] = {1.f - wz, wz};
          const float* tl = table + (size_t)lvl * (THASH * 2u);
          float f0 = 0.f, f1 = 0.f;
#pragma unroll
          for (int ci = 0; ci < 2; ++ci)
#pragma unroll
            for (int cj = 0; cj < 2; ++cj)
#pragma unroll
              for (int ck = 0; ck < 2; ++ck) {
                unsigned idx = (hx[ci] ^ hy[cj] ^ hz[ck]) & (THASH - 1u);
                const float* fp = tl + (size_t)idx * 2u;
                float wgt = wxa[ci] * wya[cj] * wza[ck];
                f0 += wgt * fp[0];
                f1 += wgt * fp[1];
              }
          h2 pk; pk[0] = (_Float16)f0; pk[1] = (_Float16)f1;
          *reinterpret_cast<h2*>(encRow + lvl * 2) = pk;
        }
      }
    }
    __syncthreads();

    // ---- layer-0 B fragments for both column groups (points 0..15 / 16..31) ----
    v16h bA[4], bB[4];
    {
      v16h e0A, e0B;
      {
        const _Float16* pA = encW + col * 32 + hh * 16;
        const _Float16* pB = encW + (16 + col) * 32 + hh * 16;
        v8h lo = *reinterpret_cast<const v8h*>(pA);
        v8h hi = *reinterpret_cast<const v8h*>(pA + 8);
        v8h lo2 = *reinterpret_cast<const v8h*>(pB);
        v8h hi2 = *reinterpret_cast<const v8h*>(pB + 8);
#pragma unroll
        for (int i = 0; i < 8; ++i) {
          e0A[i] = lo[i];  e0A[i + 8] = hi[i];
          e0B[i] = lo2[i]; e0B[i + 8] = hi2[i];
        }
      }
#pragma unroll
      for (int kt2 = 0; kt2 < 4; ++kt2) {
        v8h bf0 = bias_frag_h(biasH, 2 * kt2 + 0, hh);
        v8h bf1 = bias_frag_h(biasH, 2 * kt2 + 1, hh);
        v8h c0A = bf0, c0B = bf0, c1A = bf1, c1B = bf1;
        v16h a0 = load_w_frag(wLds + W0_OFF + (2 * kt2 + 0) * 512, lane);
        c0A = wmma_acc_h(a0, e0A, c0A);
        c0B = wmma_acc_h(a0, e0B, c0B);
        v16h a1 = load_w_frag(wLds + W0_OFF + (2 * kt2 + 1) * 512, lane);
        c1A = wmma_acc_h(a1, e0A, c1A);
        c1B = wmma_acc_h(a1, e0B, c1B);
        bA[kt2] = pack2_relu_h(c0A, c1A);
        bB[kt2] = pack2_relu_h(c0B, c1B);
      }
    }

    // ---- hidden layers 1..3: f16 accumulate, register-resident activations ----
    const int hoff[3] = {W1_OFF, W2_OFF, W3_OFF};
#pragma unroll
    for (int l = 0; l < 3; ++l) {
      const _Float16* wl = wLds + hoff[l];
      const _Float16* bl = biasH + 128 * (l + 1);
      v16h nA[4], nB[4];
#pragma unroll
      for (int kt2 = 0; kt2 < 4; ++kt2) {
        v8h bf0 = bias_frag_h(bl, 2 * kt2 + 0, hh);
        v8h bf1 = bias_frag_h(bl, 2 * kt2 + 1, hh);
        v8h c0A = bf0, c0B = bf0, c1A = bf1, c1B = bf1;
        const _Float16* f0 = wl + (2 * kt2 + 0) * 4 * 512;
        const _Float16* f1 = wl + (2 * kt2 + 1) * 4 * 512;
#pragma unroll
        for (int kt = 0; kt < 4; ++kt) {
          v16h a0 = load_w_frag(f0 + kt * 512, lane);
          c0A = wmma_acc_h(a0, bA[kt], c0A);
          c0B = wmma_acc_h(a0, bB[kt], c0B);
          v16h a1 = load_w_frag(f1 + kt * 512, lane);
          c1A = wmma_acc_h(a1, bA[kt], c1A);
          c1B = wmma_acc_h(a1, bB[kt], c1B);
        }
        nA[kt2] = pack2_relu_h(c0A, c1A);
        nB[kt2] = pack2_relu_h(c0B, c1B);
      }
#pragma unroll
      for (int i = 0; i < 4; ++i) { bA[i] = nA[i]; bB[i] = nB[i]; }
    }

    // ---- layer 4: 128 -> 4 in f32 accumulate; coalesced b128 store per point ----
    {
      v8f cA = bias_frag8(bias4, 0, hh);
      v8f cB = cA;
#pragma unroll
      for (int kt = 0; kt < 4; ++kt) {
        v16h a = load_w_frag(wLds + W4_OFF + kt * 512, lane);
        cA = wmma_acc_f(a, bA[kt], cA);
        cB = wmma_acc_f(a, bB[kt], cB);
      }
      if (hh == 0) {
        int gp = base + wave * 32 + col;
        if (gp < N) {
          v4f o; o[0] = cA[0]; o[1] = cA[1]; o[2] = cA[2]; o[3] = cA[3];
          *reinterpret_cast<v4f*>(out + (size_t)gp * 4) = o;
        }
        int gp2 = gp + 16;
        if (gp2 < N) {
          v4f o; o[0] = cB[0]; o[1] = cB[1]; o[2] = cB[2]; o[3] = cB[3];
          *reinterpret_cast<v4f*>(out + (size_t)gp2 * 4) = o;
        }
      }
    }
  }
}

extern "C" void kernel_launch(void* const* d_in, const int* in_sizes, int n_in,
                              void* d_out, int out_size, void* d_ws, size_t ws_size,
                              hipStream_t stream) {
  const float* x  = (const float*)d_in[0];
  const float* tb = (const float*)d_in[1];
  const float* W0 = (const float*)d_in[2];  const float* b0 = (const float*)d_in[3];
  const float* W1 = (const float*)d_in[4];  const float* b1 = (const float*)d_in[5];
  const float* W2 = (const float*)d_in[6];  const float* b2 = (const float*)d_in[7];
  const float* W3 = (const float*)d_in[8];  const float* b3 = (const float*)d_in[9];
  const float* W4 = (const float*)d_in[10]; const float* b4 = (const float*)d_in[11];
  float* out = (float*)d_out;

  int N = in_sizes[0] / 3;
  int nTiles = (N + TILE_PTS - 1) / TILE_PTS;
  // 4+ tiles per block: amortize the one-time weight swizzle while still
  // launching enough blocks (8192 waves) to fill every WGP at 2 blocks each.
  int blocks = nTiles < 1024 ? nTiles : 1024;

  hipFuncSetAttribute(reinterpret_cast<const void*>(ngp_fused_kernel),
                      hipFuncAttributeMaxDynamicSharedMemorySize, LDS_BYTES);
  ngp_fused_kernel<<<blocks, 256, LDS_BYTES, stream>>>(
      x, tb, W0, b0, W1, b1, W2, b2, W3, b3, W4, b4, out, N, nTiles);
}
